// _InteractionBlock_48455821033948
// MI455X (gfx1250) — compile-verified, matrix operand order checked by reading, not run on hardware
//
#include <hip/hip_runtime.h>

// ---------------------------------------------------------------------------
// Types for CDNA5 WMMA (wave32, 16x16x32 bf16 -> f32)
// ---------------------------------------------------------------------------
typedef __attribute__((ext_vector_type(16))) __bf16 bf16x16;
typedef __attribute__((ext_vector_type(8)))  __bf16 bf16x8;
typedef __attribute__((ext_vector_type(8)))  float  floatx8;

#define HDIM 128
#define NRBF 6
#define NSBF 42
#define NB   8
#define LP   136               // LDS pitch in bf16 elems (128 + 8 pad -> no bank clash)
#define SLOT_ELEMS 16384       // 128*128 elems per swizzled weight matrix
// weight slots in swizzled workspace
#define SLOT_WJI  0
#define SLOT_WKJ  1
#define SLOT_BIL0 2            // 2..9
#define SLOT_R1W1 10
#define SLOT_R1W2 11
#define SLOT_R2W1 12
#define SLOT_R2W2 13
#define SLOT_WOUT 14
#define NSLOTS    15

__device__ __forceinline__ float silu_f(float x) {
    return x / (1.0f + __expf(-x));
}

__device__ __forceinline__ floatx8 zero8() {
    floatx8 v;
#pragma unroll
    for (int i = 0; i < 8; ++i) v[i] = 0.0f;
    return v;
}

__device__ __forceinline__ floatx8 wmma_bf16(bf16x16 a, bf16x16 b, floatx8 c) {
    // D = A(16x32) * B(32x16) + C, fp32 accumulate
    return __builtin_amdgcn_wmma_f32_16x16x32_bf16(
        /*neg_a=*/false, a, /*neg_b=*/false, b,
        /*c_mod=*/(short)0, c, /*reuse_a=*/false, /*reuse_b=*/false);
}

// A fragment from a row-major bf16 row in global memory (row = this lane's M).
// 16-bit A layout: lane<16 -> K {0..7,16..23}+32*ks ; lane>=16 -> K {8..15,24..31}+32*ks
__device__ __forceinline__ bf16x16 afrag_global(const __bf16* __restrict__ row,
                                                int lane, int ks) {
    int cb = ks * 32 + ((lane & 16) ? 8 : 0);
    bf16x8 lo = *(const bf16x8*)(row + cb);
    bf16x8 hi = *(const bf16x8*)(row + cb + 16);
    bf16x16 a;
#pragma unroll
    for (int j = 0; j < 8; ++j) { a[j] = lo[j]; a[j + 8] = hi[j]; }
    return a;
}

// Raw (unscaled) A fragment from bf16 LDS tile (pitch LP).
__device__ __forceinline__ bf16x16 afrag_lds_raw(const __bf16* As, int rloc,
                                                 int lane, int ks) {
    int cb = ks * 32 + ((lane & 16) ? 8 : 0);
    const __bf16* p = As + rloc * LP + cb;
    bf16x8 lo = *(const bf16x8*)(p);
    bf16x8 hi = *(const bf16x8*)(p + 16);
    bf16x16 a;
#pragma unroll
    for (int j = 0; j < 8; ++j) { a[j] = lo[j]; a[j + 8] = hi[j]; }
    return a;
}

// Per-row scale of a bf16 fragment with packed v_pk_mul_bf16 (VOP3P).
// s2 = {scale_bf16, scale_bf16} packed in 32 bits.
__device__ __forceinline__ bf16x16 scale_frag_pk(bf16x16 a, unsigned s2) {
    union { bf16x16 h; unsigned u[8]; } in, out;
    in.h = a;
#pragma unroll
    for (int i = 0; i < 8; ++i) {
        unsigned r;
        asm("v_pk_mul_bf16 %0, %1, %2" : "=v"(r) : "v"(in.u[i]), "v"(s2));
        out.u[i] = r;
    }
    return out.h;
}

__device__ __forceinline__ unsigned splat_bf16(float s) {
    union { __bf16 h; unsigned short u; } cv;
    cv.h = (__bf16)s;
    return (unsigned)cv.u * 0x10001u;
}

// B fragment from pre-swizzled weights: fragment (ks,ns) stored lane-contiguous.
__device__ __forceinline__ bf16x16 bfrag(const __bf16* __restrict__ wslot,
                                         int lane, int ks, int ns) {
    return *(const bf16x16*)(wslot + (ks * 8 + ns) * 512 + lane * 16);
}

// Full 16x128-strip GEMM from an LDS bf16 tile: acc[ns] = A_strip @ Wslot
__device__ __forceinline__ void gemm_lds(const __bf16* As,
                                         const __bf16* __restrict__ wslot,
                                         int lane, int rloc, floatx8 acc[8]) {
#pragma unroll
    for (int i = 0; i < 8; ++i) acc[i] = zero8();
#pragma unroll
    for (int ks = 0; ks < 4; ++ks) {
        bf16x16 a = afrag_lds_raw(As, rloc, lane, ks);
#pragma unroll
        for (int ns = 0; ns < 8; ++ns)
            acc[ns] = wmma_bf16(a, bfrag(wslot, lane, ks, ns), acc[ns]);
    }
}

// ---------------------------------------------------------------------------
// K0a: m (fp32) -> bf16 copy (L2-resident gather source for triplets)
// ---------------------------------------------------------------------------
__global__ void k0_convert_m(const float* __restrict__ src,
                             __bf16* __restrict__ dst, size_t n) {
    size_t i = (size_t)blockIdx.x * blockDim.x + threadIdx.x;
    size_t stride = (size_t)gridDim.x * blockDim.x;
    for (; i < n; i += stride) dst[i] = (__bf16)src[i];
}

// ---------------------------------------------------------------------------
// K0b: pre-swizzle all weight matrices into WMMA B-fragment order (bf16).
// blockIdx.x = slot.  Fragment f=ks*8+ns holds 32 lanes x 16 bf16 contiguous.
// ---------------------------------------------------------------------------
__global__ void k0_swizzle(const float* __restrict__ Wji,
                           const float* __restrict__ Wkj,
                           const float* __restrict__ Wbil,
                           const float* __restrict__ R1w1,
                           const float* __restrict__ R1w2,
                           const float* __restrict__ R2w1,
                           const float* __restrict__ R2w2,
                           const float* __restrict__ Wout,
                           __bf16* __restrict__ wsw) {
    int slot = blockIdx.x;
    const float* src;
    int bsel = -1;
    switch (slot) {
        case SLOT_WJI:  src = Wji;  break;
        case SLOT_WKJ:  src = Wkj;  break;
        case SLOT_R1W1: src = R1w1; break;
        case SLOT_R1W2: src = R1w2; break;
        case SLOT_R2W1: src = R2w1; break;
        case SLOT_R2W2: src = R2w2; break;
        case SLOT_WOUT: src = Wout; break;
        default:        src = Wbil; bsel = slot - SLOT_BIL0; break;
    }
    for (int item = threadIdx.x; item < 1024; item += blockDim.x) {
        int f = item >> 5, lane = item & 31;
        int ks = f >> 3, ns = f & 7;
        int base = ks * 32 + ((lane & 16) ? 8 : 0);
        int n = ns * 16 + (lane & 15);
        __bf16* dst = wsw + (size_t)slot * SLOT_ELEMS + f * 512 + lane * 16;
#pragma unroll
        for (int j = 0; j < 16; ++j) {
            int k = base + (j < 8 ? j : j + 8);
            float v = (bsel < 0) ? src[k * HDIM + n]
                                 : src[k * (NB * HDIM) + bsel * HDIM + n];
            dst[j] = (__bf16)v;
        }
    }
}

// ---------------------------------------------------------------------------
// K1: aggr = silu(m @ W_ji + b_ji)    (initializes the segment-sum buffer)
// 128-edge tile per block, 8 waves, each wave = 16 rows x 128 cols
// ---------------------------------------------------------------------------
__global__ void k1_xji(const __bf16* __restrict__ mbf,
                       const __bf16* __restrict__ wsw,
                       const float* __restrict__ b_ji,
                       float* __restrict__ aggr) {
    int tid = threadIdx.x, lane = tid & 31, w = tid >> 5;
    int strip = w * 16, nc = lane & 15, mh8 = (lane & 16) ? 8 : 0;
    size_t ebase = (size_t)blockIdx.x * 128;
    const __bf16* wji = wsw + (size_t)SLOT_WJI * SLOT_ELEMS;

    floatx8 acc[8];
#pragma unroll
    for (int i = 0; i < 8; ++i) acc[i] = zero8();

    const __bf16* arow = mbf + (ebase + strip + (lane & 15)) * HDIM;
#pragma unroll
    for (int ks = 0; ks < 4; ++ks) {
        bf16x16 a = afrag_global(arow, lane, ks);
#pragma unroll
        for (int ns = 0; ns < 8; ++ns)
            acc[ns] = wmma_bf16(a, bfrag(wji, lane, ks, ns), acc[ns]);
    }
#pragma unroll
    for (int ns = 0; ns < 8; ++ns) {
        int N = ns * 16 + nc;
        float bias = __ldg(b_ji + N);
#pragma unroll
        for (int v = 0; v < 8; ++v) {
            size_t M = ebase + strip + v + mh8;
            aggr[M * HDIM + N] = silu_f(acc[ns][v] + bias);
        }
    }
}

// ---------------------------------------------------------------------------
// K2: fused triplet pipeline per 128-triplet tile:
//   gather m[kj] -> GEMM1(W_kj) -> silu*(rbf@W_rbf) -> LDS bf16 X
//   -> GEMM2: acc += (s_b * X) @ W_bil_b for b=0..7 (shared accumulators,
//      per-row scale applied with packed v_pk_mul_bf16)
//   -> atomic scatter-add into aggr[ji]
// ---------------------------------------------------------------------------
__global__ void k2_triplets(const __bf16* __restrict__ mbf,
                            const __bf16* __restrict__ wsw,
                            const float* __restrict__ rbf,
                            const float* __restrict__ sbf,
                            const int*   __restrict__ tri,
                            const float* __restrict__ W_rbf,
                            const float* __restrict__ W_sbf,
                            const float* __restrict__ b_kj,
                            float* __restrict__ aggr,
                            int T) {
    __shared__ __bf16 Xs[128 * LP];        // x_kj tile (bf16)
    __shared__ float  s_s[128 * NB];       // sbf @ W_sbf
    __shared__ float  rbf_s[128 * NRBF];   // gathered rbf rows
    __shared__ float  wrbf_s[NRBF * HDIM]; // W_rbf staged
    __shared__ float  bkj_s[HDIM];
    __shared__ int    kj_s[128];
    __shared__ int    ji_s[128];

    int tid = threadIdx.x, lane = tid & 31, w = tid >> 5;
    int strip = w * 16, nc = lane & 15, mh8 = (lane & 16) ? 8 : 0;
    int rloc = strip + (lane & 15);
    size_t t0 = (size_t)blockIdx.x * 128;

    // ---- independent staging (single barrier at the end) -----------------
    if (tid < 128) {
        int t = (int)t0 + tid;
        kj_s[tid] = tri[t];
        ji_s[tid] = tri[T + t];
        bkj_s[tid] = b_kj[tid];
    }
    // sbf projection: one thread per row, each sbf value read exactly once
    if (tid < 128) {
        float accb[NB];
#pragma unroll
        for (int b = 0; b < NB; ++b) accb[b] = 0.0f;
        const float2* s2p = (const float2*)(sbf + (t0 + tid) * NSBF);
#pragma unroll
        for (int j2 = 0; j2 < NSBF / 2; ++j2) {
            float2 sv = __ldg(s2p + j2);
            const float* wr0 = W_sbf + (2 * j2) * NB;
#pragma unroll
            for (int b = 0; b < NB; ++b)
                accb[b] += sv.x * __ldg(wr0 + b) + sv.y * __ldg(wr0 + NB + b);
        }
#pragma unroll
        for (int b = 0; b < NB; ++b) s_s[tid * NB + b] = accb[b];
    }
    // gather rbf rows of the kj edges (re-read tri directly: no dependency)
    for (int i = tid; i < 128 * NRBF; i += blockDim.x) {
        int r = i / NRBF, j = i - r * NRBF;
        int e = __ldg(tri + t0 + r);
        rbf_s[i] = __ldg(rbf + (size_t)e * NRBF + j);
    }
    for (int i = tid; i < NRBF * HDIM; i += blockDim.x)
        wrbf_s[i] = W_rbf[i];
    __syncthreads();

    // ---- GEMM1: gathered m rows @ W_kj -----------------------------------
    floatx8 acc[8];
#pragma unroll
    for (int i = 0; i < 8; ++i) acc[i] = zero8();
    const __bf16* wkj = wsw + (size_t)SLOT_WKJ * SLOT_ELEMS;
    const __bf16* arow = mbf + (size_t)kj_s[rloc] * HDIM;   // random gather (L2)
#pragma unroll
    for (int ks = 0; ks < 4; ++ks) {
        bf16x16 a = afrag_global(arow, lane, ks);
#pragma unroll
        for (int ns = 0; ns < 8; ++ns)
            acc[ns] = wmma_bf16(a, bfrag(wkj, lane, ks, ns), acc[ns]);
    }
    // epilogue: x = silu(c + b_kj) * (rbf @ W_rbf)   -> Xs (bf16)
#pragma unroll
    for (int ns = 0; ns < 8; ++ns) {
        int N = ns * 16 + nc;
        float bias = bkj_s[N];
        float w0 = wrbf_s[0 * HDIM + N], w1 = wrbf_s[1 * HDIM + N];
        float w2 = wrbf_s[2 * HDIM + N], w3 = wrbf_s[3 * HDIM + N];
        float w4 = wrbf_s[4 * HDIM + N], w5 = wrbf_s[5 * HDIM + N];
#pragma unroll
        for (int v = 0; v < 8; ++v) {
            int M = strip + v + mh8;
            const float* rr = rbf_s + M * NRBF;
            float g = rr[0] * w0 + rr[1] * w1 + rr[2] * w2 +
                      rr[3] * w3 + rr[4] * w4 + rr[5] * w5;
            Xs[M * LP + N] = (__bf16)(silu_f(acc[ns][v] + bias) * g);
        }
    }
    __syncthreads();

    // ---- GEMM2: msg = sum_b (s_b * X) @ W_bil_b  (one accumulator set) ---
    bf16x16 af[4];
#pragma unroll
    for (int ks = 0; ks < 4; ++ks)
        af[ks] = afrag_lds_raw(Xs, rloc, lane, ks);   // load X fragments once

#pragma unroll
    for (int i = 0; i < 8; ++i) acc[i] = zero8();
#pragma unroll
    for (int b = 0; b < NB; ++b) {
        const __bf16* wb = wsw + (size_t)(SLOT_BIL0 + b) * SLOT_ELEMS;
        unsigned s2 = splat_bf16(s_s[rloc * NB + b]);   // per-row scalar
#pragma unroll
        for (int ks = 0; ks < 4; ++ks) {
            bf16x16 a = scale_frag_pk(af[ks], s2);      // v_pk_mul_bf16 x8
#pragma unroll
            for (int ns = 0; ns < 8; ++ns)
                acc[ns] = wmma_bf16(a, bfrag(wb, lane, ks, ns), acc[ns]);
        }
    }

    // ---- scatter-add into aggr[ji] (L2-resident atomics) -----------------
#pragma unroll
    for (int ns = 0; ns < 8; ++ns) {
        int N = ns * 16 + nc;
#pragma unroll
        for (int v = 0; v < 8; ++v) {
            int row = strip + v + mh8;
            atomicAdd(aggr + (size_t)ji_s[row] * HDIM + N, acc[ns][v]);
        }
    }
}

// ---------------------------------------------------------------------------
// K3: edge post-processing, fully fused:
//   x = aggr; x = res1(x); x = res2(x); out = m + silu(x @ W_out + b_out)
// All 5 GEMMs via WMMA, A staged through one LDS bf16 tile.
// ---------------------------------------------------------------------------
__global__ void k3_post(const float* __restrict__ m,
                        const float* __restrict__ aggr,
                        const __bf16* __restrict__ wsw,
                        const float* __restrict__ r1b1,
                        const float* __restrict__ r1b2,
                        const float* __restrict__ r2b1,
                        const float* __restrict__ r2b2,
                        const float* __restrict__ bout,
                        float* __restrict__ out) {
    __shared__ __bf16 As[128 * LP];
    int tid = threadIdx.x, lane = tid & 31, w = tid >> 5;
    int strip = w * 16, nc = lane & 15, mh8 = (lane & 16) ? 8 : 0;
    int rloc = strip + (lane & 15);
    size_t ebase = (size_t)blockIdx.x * 128;

    floatx8 xc[8], acc[8];
#pragma unroll
    for (int ns = 0; ns < 8; ++ns) {
        int N = ns * 16 + nc;
#pragma unroll
        for (int v = 0; v < 8; ++v)
            xc[ns][v] = aggr[(ebase + strip + v + mh8) * HDIM + N];
    }

#pragma unroll
    for (int blkres = 0; blkres < 2; ++blkres) {
        const __bf16* w1 = wsw + (size_t)(blkres ? SLOT_R2W1 : SLOT_R1W1) * SLOT_ELEMS;
        const __bf16* w2 = wsw + (size_t)(blkres ? SLOT_R2W2 : SLOT_R1W2) * SLOT_ELEMS;
        const float* bb1 = blkres ? r2b1 : r1b1;
        const float* bb2 = blkres ? r2b2 : r1b2;
        // stage silu(x)
#pragma unroll
        for (int ns = 0; ns < 8; ++ns)
#pragma unroll
            for (int v = 0; v < 8; ++v)
                As[(strip + v + mh8) * LP + ns * 16 + nc] = (__bf16)silu_f(xc[ns][v]);
        __syncthreads();
        gemm_lds(As, w1, lane, rloc, acc);
        __syncthreads();
        // stage h = silu(acc + b1)
#pragma unroll
        for (int ns = 0; ns < 8; ++ns) {
            float bias = __ldg(bb1 + ns * 16 + nc);
#pragma unroll
            for (int v = 0; v < 8; ++v)
                As[(strip + v + mh8) * LP + ns * 16 + nc] =
                    (__bf16)silu_f(acc[ns][v] + bias);
        }
        __syncthreads();
        gemm_lds(As, w2, lane, rloc, acc);
        __syncthreads();
#pragma unroll
        for (int ns = 0; ns < 8; ++ns) {
            float bias = __ldg(bb2 + ns * 16 + nc);
#pragma unroll
            for (int v = 0; v < 8; ++v)
                xc[ns][v] += acc[ns][v] + bias;
        }
    }

    // out layer: stage x (no activation), GEMM W_out, silu, add skip m
#pragma unroll
    for (int ns = 0; ns < 8; ++ns)
#pragma unroll
        for (int v = 0; v < 8; ++v)
            As[(strip + v + mh8) * LP + ns * 16 + nc] = (__bf16)xc[ns][v];
    __syncthreads();
    gemm_lds(As, wsw + (size_t)SLOT_WOUT * SLOT_ELEMS, lane, rloc, acc);

#pragma unroll
    for (int ns = 0; ns < 8; ++ns) {
        int N = ns * 16 + nc;
        float bias = __ldg(bout + N);
#pragma unroll
        for (int v = 0; v < 8; ++v) {
            size_t idx = (ebase + strip + v + mh8) * HDIM + N;
            out[idx] = m[idx] + silu_f(acc[ns][v] + bias);
        }
    }
}

// ---------------------------------------------------------------------------
extern "C" void kernel_launch(void* const* d_in, const int* in_sizes, int n_in,
                              void* d_out, int out_size, void* d_ws, size_t ws_size,
                              hipStream_t stream) {
    (void)n_in; (void)out_size; (void)ws_size;
    const float* m     = (const float*)d_in[0];
    const float* rbf   = (const float*)d_in[1];
    const float* sbf   = (const float*)d_in[2];
    const int*   tri   = (const int*)  d_in[4];   // [2,T]: row0=kj, row1=ji
    const float* W_rbf = (const float*)d_in[5];
    const float* W_sbf = (const float*)d_in[6];
    const float* W_kj  = (const float*)d_in[7];
    const float* b_kj  = (const float*)d_in[8];
    const float* W_ji  = (const float*)d_in[9];
    const float* b_ji  = (const float*)d_in[10];
    const float* W_bil = (const float*)d_in[11];
    const float* r1w1  = (const float*)d_in[12];
    const float* r1b1  = (const float*)d_in[13];
    const float* r1w2  = (const float*)d_in[14];
    const float* r1b2  = (const float*)d_in[15];
    const float* r2w1  = (const float*)d_in[16];
    const float* r2b1  = (const float*)d_in[17];
    const float* r2w2  = (const float*)d_in[18];
    const float* r2b2  = (const float*)d_in[19];
    const float* W_out = (const float*)d_in[20];
    const float* b_out = (const float*)d_in[21];

    const int E = in_sizes[0] / HDIM;   // 131072
    const int T = in_sizes[2] / NSBF;   // 1048576

    char* ws = (char*)d_ws;
    float*  aggr = (float*)ws;                                      // E*H fp32
    __bf16* mbf  = (__bf16*)(ws + (size_t)E * HDIM * 4);            // E*H bf16
    __bf16* wsw  = (__bf16*)(ws + (size_t)E * HDIM * 4
                                + (size_t)E * HDIM * 2);            // swizzled W

    k0_convert_m<<<2048, 256, 0, stream>>>(m, mbf, (size_t)E * HDIM);
    k0_swizzle<<<NSLOTS, 256, 0, stream>>>(W_ji, W_kj, W_bil, r1w1, r1w2,
                                           r2w1, r2w2, W_out, wsw);
    k1_xji<<<E / 128, 256, 0, stream>>>(mbf, wsw, b_ji, aggr);
    k2_triplets<<<T / 128, 256, 0, stream>>>(mbf, wsw, rbf, sbf, tri,
                                             W_rbf, W_sbf, b_kj, aggr, T);
    k3_post<<<E / 128, 256, 0, stream>>>(m, aggr, wsw, r1b1, r1b2,
                                         r2b1, r2b2, b_out, (float*)d_out);
}
